// CrossAttention_87110526698011
// MI455X (gfx1250) — compile-verified, matrix operand order checked by reading, not run on hardware
//
#include <hip/hip_runtime.h>
#include <cstdint>
#include <cstddef>

typedef __attribute__((ext_vector_type(16))) _Float16 v16h;
typedef __attribute__((ext_vector_type(8)))  _Float16 v8h;
typedef __attribute__((ext_vector_type(8)))  float    v8f;
typedef __attribute__((ext_vector_type(4)))  int      v4i;
typedef __attribute__((ext_vector_type(8)))  int      v8i;
typedef __attribute__((ext_vector_type(4)))  unsigned int v4u;

typedef __attribute__((address_space(3))) const void lconst_void;

namespace {
constexpr int kB  = 4;
constexpr int kN  = 4096;
constexpr int kM  = 1024;
constexpr int kQD = 320;
constexpr int kCD = 768;
constexpr int kID = 512;
constexpr int kH  = 8;
constexpr int kD  = 64;
// softmax computed in log2 domain: scale = dim_head^-0.5 * log2(e)
constexpr float kScaleLog2e = 0.125f * 1.4426950408889634f;
}

union Frag16 { v16h v; v8h h[2]; };

static __device__ __forceinline__ v8f vzero8() {
  v8f r;
#pragma unroll
  for (int i = 0; i < 8; ++i) r[i] = 0.0f;
  return r;
}

static __device__ __forceinline__ v8f wmma_f16(const v16h& a, const v16h& b, const v8f& c) {
  // D = A(16x32 f16) * B(32x16 f16) + C(16x16 f32)
  return __builtin_amdgcn_wmma_f32_16x16x32_f16(false, a, false, b, (short)0, c, false, false);
}

// 16-byte async global -> LDS copy (ASYNCcnt-tracked, no VGPR data path)
static __device__ __forceinline__ void async_cp16(const _Float16* g, _Float16* l) {
  __builtin_amdgcn_global_load_async_to_lds_b128((v4i*)g, (v4i*)l, 0, 0);
}
static __device__ __forceinline__ void wait_async0() {
  __builtin_amdgcn_s_wait_asynccnt(0);
}

// raw v_exp_f32 (no libm denormal guard; flush-to-zero is fine for softmax)
static __device__ __forceinline__ float fast_exp2(float x) {
  return __builtin_amdgcn_exp2f(x);
}

// exchange with lane^16 partner via v_permlanex16_b32 (identity selects)
static __device__ __forceinline__ float swap16(float x) {
  int i = __builtin_amdgcn_permlanex16(__float_as_int(x), __float_as_int(x),
                                       0x76543210, 0xfedcba98, true, false);
  return __int_as_float(i);
}

// LDS byte offset of a generic pointer into shared memory
static __device__ __forceinline__ unsigned lds_off(const void* p) {
  return (unsigned)(size_t)(lconst_void*)p;
}

// ---- Tensor Data Mover: one-descriptor 2D tile load, global -> LDS ----
// tile_w elements (f16) per row, tile_h rows, row pitch `pitch` elements.
// pad_i/pad_a: TDM LDS pad codes (interval DWORDs = 2<<pad_i, amount DWORDs = pad_a+1)
static __device__ __forceinline__ void tdm_load_2d(const _Float16* gsrc, unsigned lds_byte,
                                                   unsigned tile_w, unsigned tile_h,
                                                   unsigned pitch, unsigned pad_i,
                                                   unsigned pad_a) {
  unsigned long long ga = (unsigned long long)(size_t)gsrc;
  v4u g0;
  g0[0] = 1u;                                   // count=1, user load descriptor
  g0[1] = lds_byte;                             // lds_addr
  g0[2] = (unsigned)(ga & 0xffffffffu);         // global_addr[31:0]
  g0[3] = (unsigned)((ga >> 32) & 0x01ffffffu)  // global_addr[56:32]
          | (2u << 30);                         // type = 2 ("image")
  v8i g1;
  g1[0] = (int)((1u << 16)                      // data_size = 1 -> 2 bytes
                | (1u << 20)                    // pad_enable
                | (pad_i << 22) | (pad_a << 25));
  g1[1] = (int)((tile_w & 0xffffu) << 16);      // tensor_dim0[15:0] (abar addr = 0)
  g1[2] = (int)(((tile_w >> 16) & 0xffffu) | ((tile_h & 0xffffu) << 16));  // dim0 hi | dim1 lo
  g1[3] = (int)(((tile_h >> 16) & 0xffffu) | ((tile_w & 0xffffu) << 16));  // dim1 hi | tile_dim0
  g1[4] = (int)(tile_h & 0xffffu);              // tile_dim1 (tile_dim2 = 0)
  g1[5] = (int)pitch;                           // tensor_dim0_stride[31:0]
  g1[6] = 0;                                    // stride0 hi | stride1 lo (unused, 1 row batch)
  g1[7] = 0;
  v4i z4 = {0, 0, 0, 0};
  v8i z8 = {0, 0, 0, 0, 0, 0, 0, 0};
  __builtin_amdgcn_tensor_load_to_lds(g0, g1, z4, z4, z8, 0);
}
static __device__ __forceinline__ void wait_tensor0() {
  __builtin_amdgcn_s_wait_tensorcnt(0);
}

// ---------------- conversion kernels ----------------
__global__ void k_cvt_f32_f16(const float* __restrict__ x, _Float16* __restrict__ y, int n) {
  int i = blockIdx.x * blockDim.x + threadIdx.x;
  int stride = gridDim.x * blockDim.x;
  for (; i < n; i += stride) y[i] = (_Float16)x[i];
}

// W is (K x Nc) row-major f32; writes Wt (Nc x K) row-major f16.
__global__ void k_cvt_transpose(const float* __restrict__ w, _Float16* __restrict__ wt,
                                int K, int Nc) {
  int i = blockIdx.x * blockDim.x + threadIdx.x;
  int n = K * Nc;
  int stride = gridDim.x * blockDim.x;
  for (; i < n; i += stride) {
    int k = i / Nc;
    int c = i - k * Nc;
    wt[(size_t)c * K + k] = (_Float16)w[i];
  }
}

// ---------------- generic WMMA GEMM ----------------
// Out = A (rows x K, f16 row-major) * Bt^T, with Bt given (Ncols x K) f16 row-major.
// Block tile: 128 (M) x 64 (N), BK = 32, 256 threads = 8 waves; wave w owns rows w*16..w*16+15.
// Double-buffered LDS, async global->LDS staging overlapped with WMMA compute.
// MODE 0: f16 out row-major [rows][Ncols]
// MODE 1: f16 out scattered to V^T layout [b][h][d][m] (rows = b*1024+m, col = h*64+d)
// MODE 2: f32 out row-major + bias
template <int MODE>
__global__ __launch_bounds__(256) void k_gemm(const _Float16* __restrict__ A,
                                              const _Float16* __restrict__ Bt,
                                              void* __restrict__ out,
                                              const float* __restrict__ bias,
                                              int rows, int K, int Ncols) {
  (void)rows;
  __shared__ _Float16 sA[2][128][40];
  __shared__ _Float16 sB[2][64][40];

  const int tid  = threadIdx.x;
  const int wave = tid >> 5;
  const int lane = tid & 31;
  const int c    = lane & 15;   // fragment row/col index within 16
  const int hh   = lane >> 4;   // half-wave select

  const int row0 = blockIdx.x * 128;
  const int col0 = blockIdx.y * 64;

  // issue async staging of one 128x32 A tile + 64x32 B tile into buffer `buf`
  auto stage = [&](int k0, int buf) {
#pragma unroll
    for (int s = 0; s < 2; ++s) {
      int t = tid + s * 256;
      int r = t >> 2, k8 = (t & 3) * 8;
      async_cp16(A + (size_t)(row0 + r) * K + k0 + k8, &sA[buf][r][k8]);
    }
    {
      int r = tid >> 2, k8 = (tid & 3) * 8;
      async_cp16(Bt + (size_t)(col0 + r) * K + k0 + k8, &sB[buf][r][k8]);
    }
  };

  v8f acc[4];
#pragma unroll
  for (int t = 0; t < 4; ++t) acc[t] = vzero8();

  const int nk = K >> 5;
  stage(0, 0);
  for (int it = 0; it < nk; ++it) {
    const int cur = it & 1;
    wait_async0();      // my async writes into `cur` are in LDS
    __syncthreads();    // everyone's are
    if (it + 1 < nk) stage((it + 1) << 5, cur ^ 1);  // overlap next tile with compute

    Frag16 af;
    af.h[0] = *reinterpret_cast<const v8h*>(&sA[cur][wave * 16 + c][hh * 8]);
    af.h[1] = *reinterpret_cast<const v8h*>(&sA[cur][wave * 16 + c][16 + hh * 8]);
#pragma unroll
    for (int nt = 0; nt < 4; ++nt) {
      Frag16 bf;
      bf.h[0] = *reinterpret_cast<const v8h*>(&sB[cur][nt * 16 + c][hh * 8]);
      bf.h[1] = *reinterpret_cast<const v8h*>(&sB[cur][nt * 16 + c][16 + hh * 8]);
      acc[nt] = wmma_f16(af.v, bf.v, acc[nt]);
    }
  }

  // epilogue: C layout -> lane holds col c, VGPR i holds local row i + 8*hh
#pragma unroll
  for (int nt = 0; nt < 4; ++nt) {
#pragma unroll
    for (int i = 0; i < 8; ++i) {
      int row = row0 + wave * 16 + 8 * hh + i;
      int col = col0 + nt * 16 + c;
      float v = acc[nt][i];
      if (MODE == 0) {
        reinterpret_cast<_Float16*>(out)[(size_t)row * Ncols + col] = (_Float16)v;
      } else if (MODE == 1) {
        int bb   = row >> 10;        // kM == 1024
        int m    = row & 1023;
        int head = col >> 6;         // kD == 64
        int d    = col & 63;
        reinterpret_cast<_Float16*>(out)[(((size_t)bb * kH + head) * kD + d) * kM + m] =
            (_Float16)v;
      } else {
        reinterpret_cast<float*>(out)[(size_t)row * Ncols + col] = v + bias[col];
      }
    }
  }
}

// ---------------- fused flash attention ----------------
// One block = (b, head, 64 Q rows); 4 waves, each owning 16 Q rows.
// Computes S^T = K_chunk * Q^T so softmax reductions are lane-local (+ one permlanex16),
// and the per-lane P^T values feed the O^T WMMA directly as B-fragments (no LDS transpose).
// K/V chunks double-buffered in LDS, staged by the Tensor Data Mover (one descriptor per
// tile, issued by wave 0, TDM hardware applies the +8-half LDS pad for bank conflicts).
__global__ __launch_bounds__(128) void k_attn(const _Float16* __restrict__ qh,
                                              const _Float16* __restrict__ kh,
                                              const _Float16* __restrict__ vT,
                                              _Float16* __restrict__ oh) {
  __shared__ _Float16 sK[2][32][72];   // K chunk: 32 m-rows x 64 d (+8 pad)
  __shared__ _Float16 sV[2][64][40];   // V^T chunk: 64 d-rows x 32 m (+8 pad)

  const int tid  = threadIdx.x;
  const int wave = tid >> 5;
  const int lane = tid & 31;
  const int c    = lane & 15;
  const int hh   = lane >> 4;

  const int n0   = blockIdx.x * 64 + wave * 16;
  const int head = blockIdx.y;
  const int b    = blockIdx.z;

  const _Float16* qrow  = qh + ((size_t)b * kN + n0 + c) * kID + head * kD;
  const _Float16* kbase = kh + (size_t)b * kM * kID + head * kD;
  const _Float16* vbase = vT + ((size_t)(b * kH + head)) * kD * kM;

  // TDM staging: one descriptor per tile; wave 0 issues for the whole block.
  // K tile: 64 halfs/row (32 DWORDs -> pad_i=4), pad 4 DWORDs (pad_a=3) => stride 72.
  // V tile: 32 halfs/row (16 DWORDs -> pad_i=3), pad 4 DWORDs (pad_a=3) => stride 40.
  auto stage = [&](int mc, int buf) {
    tdm_load_2d(kbase + (size_t)mc * kID, lds_off(&sK[buf][0][0]), 64, 32, kID, 4, 3);
    tdm_load_2d(vbase + mc, lds_off(&sV[buf][0][0]), 32, 64, kM, 3, 3);
  };

  // Q as B-fragment of S^T (lane c -> Q row n0+c), pre-scaled by scale*log2(e) so the
  // WMMA outputs are already in the log2 softmax domain.
  Frag16 qb0, qb1;
  qb0.h[0] = *reinterpret_cast<const v8h*>(qrow + hh * 8);
  qb0.h[1] = *reinterpret_cast<const v8h*>(qrow + 16 + hh * 8);
  qb1.h[0] = *reinterpret_cast<const v8h*>(qrow + 32 + hh * 8);
  qb1.h[1] = *reinterpret_cast<const v8h*>(qrow + 48 + hh * 8);
  const _Float16 qs = (_Float16)kScaleLog2e;
#pragma unroll
  for (int i = 0; i < 16; ++i) {
    qb0.v[i] *= qs;
    qb1.v[i] *= qs;
  }

  v8f ot[4];  // O^T accumulators: 4 d-tiles x (16x16)
#pragma unroll
  for (int t = 0; t < 4; ++t) ot[t] = vzero8();
  float m_run = -1e30f, l_run = 0.0f;

  if (wave == 0) stage(0, 0);
  for (int it = 0; it < (kM >> 5); ++it) {
    const int cur = it & 1;
    const int mcn = (it + 1) << 5;
    if (wave == 0) wait_tensor0();  // wave 0's TDM writes into `cur` complete
    __syncthreads();                // published to all waves
    if (mcn < kM && wave == 0) stage(mcn, cur ^ 1);  // overlap next chunk with compute

    // S^T tiles: m-local 0..15 (s0) and 16..31 (s1); two WMMAs each (d=0..31, 32..63)
    Frag16 ka;
    v8f s0 = vzero8(), s1 = vzero8();
    ka.h[0] = *reinterpret_cast<const v8h*>(&sK[cur][c][hh * 8]);
    ka.h[1] = *reinterpret_cast<const v8h*>(&sK[cur][c][16 + hh * 8]);
    s0 = wmma_f16(ka.v, qb0.v, s0);
    ka.h[0] = *reinterpret_cast<const v8h*>(&sK[cur][c][32 + hh * 8]);
    ka.h[1] = *reinterpret_cast<const v8h*>(&sK[cur][c][48 + hh * 8]);
    s0 = wmma_f16(ka.v, qb1.v, s0);
    ka.h[0] = *reinterpret_cast<const v8h*>(&sK[cur][16 + c][hh * 8]);
    ka.h[1] = *reinterpret_cast<const v8h*>(&sK[cur][16 + c][16 + hh * 8]);
    s1 = wmma_f16(ka.v, qb0.v, s1);
    ka.h[0] = *reinterpret_cast<const v8h*>(&sK[cur][16 + c][32 + hh * 8]);
    ka.h[1] = *reinterpret_cast<const v8h*>(&sK[cur][16 + c][48 + hh * 8]);
    s1 = wmma_f16(ka.v, qb1.v, s1);

    // online softmax (already log2-scaled); lane holds one Q column, 16 m values
    float smax = -1e30f;
#pragma unroll
    for (int i = 0; i < 8; ++i) smax = fmaxf(smax, fmaxf(s0[i], s1[i]));
    smax = fmaxf(smax, swap16(smax));
    float mnew  = fmaxf(m_run, smax);
    float alpha = fast_exp2(m_run - mnew);
    float psum  = 0.0f;
    Frag16 pb;  // P^T B-fragment built directly in registers
#pragma unroll
    for (int i = 0; i < 8; ++i) {
      float p0 = fast_exp2(s0[i] - mnew);
      float p1 = fast_exp2(s1[i] - mnew);
      psum += p0 + p1;
      pb.v[i]     = (_Float16)p0;
      pb.v[8 + i] = (_Float16)p1;
    }
    psum += swap16(psum);
    l_run = l_run * alpha + psum;
    m_run = mnew;

#pragma unroll
    for (int t = 0; t < 4; ++t)
#pragma unroll
      for (int i = 0; i < 8; ++i) ot[t][i] *= alpha;

    // O^T += V^T_tile(16d x 32m) * P^T(32m x 16n)
#pragma unroll
    for (int dt = 0; dt < 4; ++dt) {
      Frag16 va;
      va.h[0] = *reinterpret_cast<const v8h*>(&sV[cur][dt * 16 + c][hh * 8]);
      va.h[1] = *reinterpret_cast<const v8h*>(&sV[cur][dt * 16 + c][16 + hh * 8]);
      ot[dt] = wmma_f16(va.v, pb.v, ot[dt]);
    }
  }

  // epilogue: O^T C-layout -> lane col n = n0+c, VGPR i -> d = dt*16 + 8*hh + i
  const float inv = 1.0f / l_run;
  _Float16* orow = oh + ((size_t)b * kN + n0 + c) * kID + head * kD;
#pragma unroll
  for (int dt = 0; dt < 4; ++dt)
#pragma unroll
    for (int i = 0; i < 8; ++i)
      orow[dt * 16 + 8 * hh + i] = (_Float16)(ot[dt][i] * inv);
}

// ---------------- host launch ----------------
extern "C" void kernel_launch(void* const* d_in, const int* in_sizes, int n_in,
                              void* d_out, int out_size, void* d_ws, size_t ws_size,
                              hipStream_t stream) {
  (void)in_sizes; (void)n_in; (void)out_size; (void)ws_size;
  const float* x   = (const float*)d_in[0];
  const float* ctx = (const float*)d_in[1];
  const float* Wq  = (const float*)d_in[2];
  const float* Wk  = (const float*)d_in[3];
  const float* Wv  = (const float*)d_in[4];
  const float* Wo  = (const float*)d_in[5];
  const float* bo  = (const float*)d_in[6];

  _Float16* ws = (_Float16*)d_ws;
  size_t off = 0;
  auto alloc = [&](size_t nelem) { _Float16* p = ws + off; off += nelem; return p; };
  _Float16* xh  = alloc((size_t)kB * kN * kQD);     // x in f16
  _Float16* chh = alloc((size_t)kB * kM * kCD);     // context in f16
  _Float16* wqT = alloc((size_t)kID * kQD);         // Wq^T
  _Float16* wkT = alloc((size_t)kID * kCD);         // Wk^T
  _Float16* wvT = alloc((size_t)kID * kCD);         // Wv^T
  _Float16* woT = alloc((size_t)kQD * kID);         // Wo^T
  _Float16* qh  = alloc((size_t)kB * kN * kID);     // q  [b][n][h*64+d]
  _Float16* khp = alloc((size_t)kB * kM * kID);     // k  [b][m][h*64+d]
  _Float16* vTp = alloc((size_t)kB * kH * kD * kM); // v^T [b][h][d][m]
  _Float16* oh  = alloc((size_t)kB * kN * kID);     // attention output f16

  k_cvt_f32_f16<<<512, 256, 0, stream>>>(x, xh, kB * kN * kQD);
  k_cvt_f32_f16<<<512, 256, 0, stream>>>(ctx, chh, kB * kM * kCD);
  k_cvt_transpose<<<256, 256, 0, stream>>>(Wq, wqT, kQD, kID);
  k_cvt_transpose<<<256, 256, 0, stream>>>(Wk, wkT, kCD, kID);
  k_cvt_transpose<<<256, 256, 0, stream>>>(Wv, wvT, kCD, kID);
  k_cvt_transpose<<<256, 256, 0, stream>>>(Wo, woT, kID, kQD);

  // q = x @ Wq ; k = ctx @ Wk ; v = ctx @ Wv (stored transposed per head)
  k_gemm<0><<<dim3((kB * kN) / 128, kID / 64), 256, 0, stream>>>(
      xh, wqT, qh, nullptr, kB * kN, kQD, kID);
  k_gemm<0><<<dim3((kB * kM) / 128, kID / 64), 256, 0, stream>>>(
      chh, wkT, khp, nullptr, kB * kM, kCD, kID);
  k_gemm<1><<<dim3((kB * kM) / 128, kID / 64), 256, 0, stream>>>(
      chh, wvT, vTp, nullptr, kB * kM, kCD, kID);

  // fused softmax(q k^T / sqrt(d)) v
  k_attn<<<dim3(kN / 64, kH, kB), 128, 0, stream>>>(qh, khp, vTp, oh);

  // out = o @ Wo + bo (f32 output)
  k_gemm<2><<<dim3((kB * kN) / 128, kQD / 64), 256, 0, stream>>>(
      oh, woT, d_out, bo, kB * kN, kID, kQD);
}
